// GERF_BRA_44710609551934
// MI455X (gfx1250) — matrix operand order repaired and checked by reading, not compiled, not run on hardware
//
#include <hip/hip_runtime.h>
#include <hip/hip_bf16.h>
#include <math.h>

typedef __attribute__((ext_vector_type(16))) _Float16 v16h;
typedef __attribute__((ext_vector_type(8)))  _Float16 v8h;
typedef __attribute__((ext_vector_type(8)))  float    v8f;

#define DEVFN static __device__ __forceinline__

constexpr int CB = 8;          // batch
constexpr int CH = 56, CW = 56;
constexpr int CC = 256;
constexpr int CS = 64;         // regions
constexpr int NT = 49;         // tokens per region
constexpr int NTOK = CB * CS * NT;   // 25088 rows
constexpr float SCALE = 0.1767766952966369f;  // (256/8)^-0.5

DEVFN v8f wmma16(v16h a, v16h b, v8f c) {
  return __builtin_amdgcn_wmma_f32_16x16x32_f16(false, a, false, b, (short)0, c,
                                                false, false);
}

// CDNA5 async global->LDS copy (16B), tracked by ASYNCcnt.
// LDS operand: low 32 bits of the flat shared address == LDS byte offset.
DEVFN void async_b128(const void* g, void* l) {
  unsigned lofs = (unsigned)(uintptr_t)l;
  asm volatile("global_load_async_to_lds_b128 %0, %1, off"
               :: "v"(lofs), "v"(g) : "memory");
}
template <int N>
DEVFN void wait_async() {
  asm volatile("s_wait_asynccnt %0" :: "i"(N) : "memory");
}

// ---------------- kernel 0a: f32 -> f16 bulk convert ----------------
__global__ __launch_bounds__(256) void k_cvt(const float* __restrict__ src,
                                             _Float16* __restrict__ dst) {
  size_t i = ((size_t)blockIdx.x * 256 + threadIdx.x) * 8;
  v8f v = *(const v8f*)&src[i];
  v8h hv;
  #pragma unroll
  for (int e = 0; e < 8; ++e) hv[e] = (_Float16)v[e];
  *(v8h*)&dst[i] = hv;
}

// ---------------- kernel 0b: weight transpose f32[K][N] -> f16[N][K] ----------------
__global__ __launch_bounds__(256) void k_wtrans(const float* __restrict__ w,
                                                _Float16* __restrict__ wt,
                                                int K, int N) {
  int o = blockIdx.x * 256 + threadIdx.x;      // o = j*K + k
  int j = o / K, k = o % K;
  wt[o] = (_Float16)w[(size_t)k * N + j];
}

// ---------------- kernel 1: region mean ----------------
__global__ __launch_bounds__(256) void k_region_mean(const float* __restrict__ x,
                                                     float* __restrict__ xr) {
  int bs = blockIdx.x;
  int b = bs >> 6, s = bs & 63;
  int c = threadIdx.x;
  int y0 = (s >> 3) * 7, x0 = (s & 7) * 7;
  float sum = 0.f;
  for (int iy = 0; iy < 7; ++iy)
    for (int ix = 0; ix < 7; ++ix)
      sum += x[(((size_t)b * CH + (y0 + iy)) * CW + (x0 + ix)) * CC + c];
  xr[(size_t)bs * CC + c] = sum * (1.0f / 49.0f);
}

// ---------------- kernel 2: region qkv (q_r, k_r only) ----------------
__global__ __launch_bounds__(256) void k_region_qkv(const float* __restrict__ xr,
                                                    const float* __restrict__ w,
                                                    const float* __restrict__ bq,
                                                    float* __restrict__ qr,
                                                    float* __restrict__ kr) {
  int o = blockIdx.x * 256 + threadIdx.x;   // 512 rows x 512 cols
  int row = o >> 9, col = o & 511;
  const float* xrow = xr + (size_t)row * CC;
  float acc = bq[col];
  for (int k = 0; k < CC; ++k) acc += xrow[k] * w[(size_t)k * 768 + col];
  if (col < 256) qr[(size_t)row * CC + col] = acc;
  else           kr[(size_t)row * CC + (col - 256)] = acc;
}

// ---------------- kernel 3: routing (topk + gaussian gate) ----------------
__global__ __launch_bounds__(64) void k_route(const float* __restrict__ qr,
                                              const float* __restrict__ kr,
                                              const float* __restrict__ wg1,
                                              const float* __restrict__ bg1,
                                              const float* __restrict__ wg2,
                                              const float* __restrict__ bg2,
                                              int* __restrict__ topk,
                                              float* __restrict__ gwt) {
  __shared__ float qsr[256];
  __shared__ float ar[64];
  __shared__ float hbuf[64];
  __shared__ float gbuf[64];
  __shared__ int   idxs[4];
  __shared__ float sig[2];
  int bs = blockIdx.x, tid = threadIdx.x;
  int b = bs >> 6, s = bs & 63;
  for (int i = tid; i < 256; i += 64) qsr[i] = qr[(size_t)bs * CC + i];
  __syncthreads();
  {
    const float* krow = kr + (size_t)(b * 64 + tid) * CC;
    float a = 0.f;
    for (int k = 0; k < CC; ++k) a += qsr[k] * krow[k];
    ar[tid] = a * SCALE;
  }
  {
    float h = bg1[tid];
    for (int k = 0; k < CC; ++k) h += qsr[k] * wg1[(size_t)k * 64 + tid];
    hbuf[tid] = h > 0.f ? h : 0.f;
  }
  __syncthreads();
  if (tid == 0) {
    unsigned long long used = 0ull;
    for (int r = 0; r < 4; ++r) {
      int bi = 0; float bv = -3.4e38f;
      for (int z = 0; z < 64; ++z)
        if (!((used >> z) & 1ull) && ar[z] > bv) { bv = ar[z]; bi = z; }
      used |= (1ull << bi);
      idxs[r] = bi;
    }
    float g0 = bg2[0], g1 = bg2[1];
    for (int j = 0; j < 64; ++j) { g0 += hbuf[j] * wg2[j * 2 + 0]; g1 += hbuf[j] * wg2[j * 2 + 1]; }
    float s0 = g0 > 0.f ? g0 + log1pf(expf(-g0)) : log1pf(expf(g0));
    float s1 = g1 > 0.f ? g1 + log1pf(expf(-g1)) : log1pf(expf(g1));
    sig[0] = s0 + 0.5f; sig[1] = s1 + 0.5f;
  }
  __syncthreads();
  {
    float dx = (float)(tid & 7) - (float)(s & 7);
    float dy = (float)(tid >> 3) - (float)(s >> 3);
    float z0 = dx / sig[0], z1 = dy / sig[1];
    gbuf[tid] = expf(-0.5f * (z0 * z0 + z1 * z1));
  }
  __syncthreads();
  if (tid == 0) {
    float sum = 0.f;
    for (int z = 0; z < 64; ++z) sum += gbuf[z];
    sum += 1e-6f;
    for (int r = 0; r < 4; ++r) {
      topk[bs * 4 + r] = idxs[r];
      gwt[bs * 4 + r]  = gbuf[idxs[r]] / sum;
    }
  }
}

// ---------------- kernel 4: token QKV GEMM (WMMA + double-buffered async staging) ----------------
__global__ __launch_bounds__(256) void k_qkv_gemm(const _Float16* __restrict__ xh,
                                                  const _Float16* __restrict__ wt,   // [768][256] f16
                                                  const float* __restrict__ bias,
                                                  _Float16* __restrict__ qh,
                                                  _Float16* __restrict__ kh,
                                                  _Float16* __restrict__ vh) {
  __shared__ __align__(16) _Float16 As[2][64][32];
  __shared__ __align__(16) _Float16 Bt[2][64][32];
  int tid = threadIdx.x;
  int wave = tid >> 5, lane = tid & 31;
  int wm = wave & 3, wn = wave >> 2;
  int l15 = lane & 15, h = lane >> 4;
  int rowbase = blockIdx.x * 64;
  int colbase = blockIdx.y * 64;
  int m8 = tid >> 2, kl8 = (tid & 3) * 8;    // one 16B chunk per thread per tile
  const _Float16* asrc = xh + (size_t)(rowbase + m8) * 256 + kl8;
  const _Float16* bsrc = wt + (size_t)(colbase + m8) * 256 + kl8;
  v8f acc0 = {}; v8f acc1 = {};
  // prologue: stage tile 0
  async_b128(asrc, &As[0][m8][kl8]);
  async_b128(bsrc, &Bt[0][m8][kl8]);
  for (int kk = 0; kk < 8; ++kk) {
    int cur = kk & 1;
    if (kk < 7) {
      async_b128(asrc + (kk + 1) * 32, &As[cur ^ 1][m8][kl8]);
      async_b128(bsrc + (kk + 1) * 32, &Bt[cur ^ 1][m8][kl8]);
      if (kk < 6) {
        __builtin_prefetch(asrc + (kk + 2) * 32, 0, 0);
        __builtin_prefetch(bsrc + (kk + 2) * 32, 0, 0);
      }
      wait_async<2>();     // tile kk's 2 copies retired; kk+1 still in flight
    } else {
      wait_async<0>();
    }
    __syncthreads();
    v16h a;
    ((v8h*)&a)[0] = *(const v8h*)&As[cur][wm * 16 + l15][h * 8];
    ((v8h*)&a)[1] = *(const v8h*)&As[cur][wm * 16 + l15][16 + h * 8];
    v16h b0, b1;
    ((v8h*)&b0)[0] = *(const v8h*)&Bt[cur][wn * 32 + l15][h * 8];
    ((v8h*)&b0)[1] = *(const v8h*)&Bt[cur][wn * 32 + l15][16 + h * 8];
    ((v8h*)&b1)[0] = *(const v8h*)&Bt[cur][wn * 32 + 16 + l15][h * 8];
    ((v8h*)&b1)[1] = *(const v8h*)&Bt[cur][wn * 32 + 16 + l15][16 + h * 8];
    acc0 = wmma16(a, b0, acc0);
    acc1 = wmma16(a, b1, acc1);
    __syncthreads();
  }
  #pragma unroll
  for (int sub = 0; sub < 2; ++sub) {
    v8f A = sub ? acc1 : acc0;
    int j = colbase + wn * 32 + sub * 16 + l15;
    _Float16* dst = (j < 256) ? qh : ((j < 512) ? kh : vh);
    int jc = j & 255;
    float bj = bias[j];
    #pragma unroll
    for (int r = 0; r < 8; ++r) {
      int R = rowbase + wm * 16 + h * 8 + r;
      int bidx = R / 3136, rem = R % 3136;
      int y = rem / 56, xx = rem % 56;
      int s = (y / 7) * 8 + xx / 7;
      int t = (y % 7) * 7 + xx % 7;
      dst[((size_t)((bidx * 64 + s) * 49 + t)) * CC + jc] = (_Float16)(A[r] + bj);
    }
  }
}

// ---------------- kernel 5: routed attention per (b,s), WMMA ----------------
__global__ __launch_bounds__(128) void k_attn(const _Float16* __restrict__ qh,
                                              const _Float16* __restrict__ kh,
                                              const _Float16* __restrict__ vh,
                                              const int* __restrict__ topk,
                                              const float* __restrict__ gwt,
                                              _Float16* __restrict__ aoh) {
  __shared__ __align__(16) _Float16 probs[49 * 256];   // ~24.5 KB
  __shared__ __align__(16) _Float16 vt[256][64];       // 32 KB (64-key chunk, transposed)
  __shared__ int idxs[4];
  __shared__ float gws[4];
  int bs = blockIdx.x;
  int b = bs >> 6;
  int tid = threadIdx.x, wave = tid >> 5, lane = tid & 31;
  int l15 = lane & 15, h = lane >> 4;
  if (tid < 4) { idxs[tid] = topk[bs * 4 + tid]; gws[tid] = gwt[bs * 4 + tid]; }
  __syncthreads();
  int mt = wave;                       // each wave owns one 16-row M-tile
  int mrow = mt * 16 + l15;
  size_t qbase = (size_t)bs * 49 * 256;

  const _Float16* kp[16];
  float gsc[16];
  bool  kval[16];
  #pragma unroll
  for (int nt = 0; nt < 16; ++nt) {
    int n = nt * 16 + l15;
    int r = n >> 6, tt = n & 63;
    int ttc = tt < 49 ? tt : 48;
    kp[nt] = kh + ((size_t)((b * 64 + idxs[r]) * 49 + ttc)) * 256 + h * 16;
    gsc[nt] = gws[r] * SCALE;    // k_g = k * gwt  ->  logits scale linearly
    kval[nt] = (tt < 49);
  }

  // ---- phase 1: logits = q @ k_g^T ----
  v8f acc[16];
  #pragma unroll
  for (int i = 0; i < 16; ++i) acc[i] = {};
  for (int ks = 0; ks < 8; ++ks) {
    v16h a;
    if (mrow < 49) {
      a = *(const v16h*)(qh + qbase + (size_t)mrow * 256 + ks * 32 + h * 16);
    } else {
      #pragma unroll
      for (int e = 0; e < 16; ++e) a[e] = (_Float16)0.0f;
    }
    #pragma unroll
    for (int nt = 0; nt < 16; ++nt) {
      v16h bb = *(const v16h*)(kp[nt] + ks * 32);
      acc[nt] = wmma16(a, bb, acc[nt]);
    }
  }
  #pragma unroll
  for (int nt = 0; nt < 16; ++nt) {
    float sc = gsc[nt]; bool v = kval[nt];
    #pragma unroll
    for (int r = 0; r < 8; ++r) acc[nt][r] = v ? acc[nt][r] * sc : -1e30f;
  }
  // ---- phase 2: softmax in registers ----
  #pragma unroll
  for (int r = 0; r < 8; ++r) {
    float mx = -1e30f;
    #pragma unroll
    for (int nt = 0; nt < 16; ++nt) mx = fmaxf(mx, acc[nt][r]);
    for (int o = 1; o < 16; o <<= 1) mx = fmaxf(mx, __shfl_xor(mx, o, 32));
    float sum = 0.f;
    #pragma unroll
    for (int nt = 0; nt < 16; ++nt) {
      float e = expf(acc[nt][r] - mx);
      acc[nt][r] = e; sum += e;
    }
    for (int o = 1; o < 16; o <<= 1) sum += __shfl_xor(sum, o, 32);
    float inv = 1.0f / sum;
    #pragma unroll
    for (int nt = 0; nt < 16; ++nt) acc[nt][r] *= inv;
  }
  #pragma unroll
  for (int nt = 0; nt < 16; ++nt) {
    int n = nt * 16 + l15;
    #pragma unroll
    for (int r = 0; r < 8; ++r) {
      int m = mt * 16 + h * 8 + r;
      if (m < 49) probs[m * 256 + n] = (_Float16)acc[nt][r];
    }
  }

  // ---- phase 3: out = probs @ v_g ----
  v8f occ[16];
  #pragma unroll
  for (int i = 0; i < 16; ++i) occ[i] = {};
  for (int c2 = 0; c2 < 4; ++c2) {         // 64-key chunks
    __syncthreads();
    {
      int kl = tid & 63, ch = tid >> 6;
      int key = c2 * 64 + kl;
      int r = key >> 6, tt = key & 63;
      int ttc = tt < 49 ? tt : 48;
      const _Float16* src = vh + ((size_t)((b * 64 + idxs[r]) * 49 + ttc)) * 256 + ch * 128;
      float g = gws[r];
      for (int cc = 0; cc < 128; cc += 8) {
        v8h vv = *(const v8h*)(src + cc);
        #pragma unroll
        for (int e = 0; e < 8; ++e)
          vt[ch * 128 + cc + e][kl] = (_Float16)((float)vv[e] * g);
      }
    }
    __syncthreads();
    for (int ks = 0; ks < 2; ++ks) {
      int kbase = c2 * 64 + ks * 32;
      v16h a;
      if (mrow < 49) {
        a = *(const v16h*)&probs[mrow * 256 + kbase + h * 16];
      } else {
        #pragma unroll
        for (int e = 0; e < 16; ++e) a[e] = (_Float16)0.0f;
      }
      #pragma unroll
      for (int nt = 0; nt < 16; ++nt) {
        v16h bb = *(const v16h*)&vt[nt * 16 + l15][ks * 32 + h * 16];
        occ[nt] = wmma16(a, bb, occ[nt]);
      }
    }
  }
  #pragma unroll
  for (int nt = 0; nt < 16; ++nt) {
    int n = nt * 16 + l15;
    #pragma unroll
    for (int r = 0; r < 8; ++r) {
      int m = mt * 16 + h * 8 + r;
      if (m < 49) aoh[qbase + (size_t)m * 256 + n] = (_Float16)occ[nt][r];
    }
  }
}

// ---------------- kernel 6: depthwise 3x3 conv on v (+= into aoh, f16 in place) ----------------
__global__ __launch_bounds__(256) void k_dwconv(const _Float16* __restrict__ vh,
                                                const float* __restrict__ kern,
                                                const float* __restrict__ kb,
                                                _Float16* __restrict__ aoh) {
  int bs = blockIdx.x;
  int c = threadIdx.x;
  size_t base = (size_t)bs * 49 * 256;
  float kr[9];
  #pragma unroll
  for (int t = 0; t < 9; ++t) kr[t] = kern[(size_t)t * 256 + c];
  float bias = kb[c];
  for (int p = 0; p < 49; ++p) {
    int iy = p / 7, ix = p % 7;
    float sum = bias;
    #pragma unroll
    for (int dy = -1; dy <= 1; ++dy) {
      int yy = iy + dy; if (yy < 0 || yy >= 7) continue;
      #pragma unroll
      for (int dx = -1; dx <= 1; ++dx) {
        int xx = ix + dx; if (xx < 0 || xx >= 7) continue;
        sum += (float)vh[base + (size_t)(yy * 7 + xx) * 256 + c] * kr[(dy + 1) * 3 + (dx + 1)];
      }
    }
    size_t o = base + (size_t)p * 256 + c;
    aoh[o] = (_Float16)((float)aoh[o] + sum);
  }
}

// ---------------- kernel 7: projection GEMM (WMMA + double-buffered async), window->raw ----------------
__global__ __launch_bounds__(256) void k_proj(const _Float16* __restrict__ aoh,
                                              const _Float16* __restrict__ wt,  // [256][256] f16
                                              const float* __restrict__ bias,
                                              float* __restrict__ out) {
  __shared__ __align__(16) _Float16 As[2][64][32];
  __shared__ __align__(16) _Float16 Bt[2][64][32];
  int tid = threadIdx.x;
  int wave = tid >> 5, lane = tid & 31;
  int wm = wave & 3, wn = wave >> 2;
  int l15 = lane & 15, h = lane >> 4;
  int rowbase = blockIdx.x * 64;
  int colbase = blockIdx.y * 64;
  int m8 = tid >> 2, kl8 = (tid & 3) * 8;
  const _Float16* asrc = aoh + (size_t)(rowbase + m8) * 256 + kl8;
  const _Float16* bsrc = wt  + (size_t)(colbase + m8) * 256 + kl8;
  v8f acc0 = {}; v8f acc1 = {};
  async_b128(asrc, &As[0][m8][kl8]);
  async_b128(bsrc, &Bt[0][m8][kl8]);
  for (int kk = 0; kk < 8; ++kk) {
    int cur = kk & 1;
    if (kk < 7) {
      async_b128(asrc + (kk + 1) * 32, &As[cur ^ 1][m8][kl8]);
      async_b128(bsrc + (kk + 1) * 32, &Bt[cur ^ 1][m8][kl8]);
      if (kk < 6) {
        __builtin_prefetch(asrc + (kk + 2) * 32, 0, 0);
        __builtin_prefetch(bsrc + (kk + 2) * 32, 0, 0);
      }
      wait_async<2>();
    } else {
      wait_async<0>();
    }
    __syncthreads();
    v16h a;
    ((v8h*)&a)[0] = *(const v8h*)&As[cur][wm * 16 + l15][h * 8];
    ((v8h*)&a)[1] = *(const v8h*)&As[cur][wm * 16 + l15][16 + h * 8];
    v16h b0, b1;
    ((v8h*)&b0)[0] = *(const v8h*)&Bt[cur][wn * 32 + l15][h * 8];
    ((v8h*)&b0)[1] = *(const v8h*)&Bt[cur][wn * 32 + l15][16 + h * 8];
    ((v8h*)&b1)[0] = *(const v8h*)&Bt[cur][wn * 32 + 16 + l15][h * 8];
    ((v8h*)&b1)[1] = *(const v8h*)&Bt[cur][wn * 32 + 16 + l15][16 + h * 8];
    acc0 = wmma16(a, b0, acc0);
    acc1 = wmma16(a, b1, acc1);
    __syncthreads();
  }
  #pragma unroll
  for (int sub = 0; sub < 2; ++sub) {
    v8f A = sub ? acc1 : acc0;
    int j = colbase + wn * 32 + sub * 16 + l15;
    float bj = bias[j];
    #pragma unroll
    for (int r = 0; r < 8; ++r) {
      int R = rowbase + wm * 16 + h * 8 + r;       // window-layout row
      int bidx = R / 3136, rem = R % 3136;
      int s = rem / 49, t = rem % 49;
      int y = (s >> 3) * 7 + t / 7;
      int xx = (s & 7) * 7 + t % 7;
      out[((size_t)(bidx * 56 + y) * 56 + xx) * CC + j] = A[r] + bj;
    }
  }
}

// ---------------- launcher ----------------
extern "C" void kernel_launch(void* const* d_in, const int* in_sizes, int n_in,
                              void* d_out, int out_size, void* d_ws, size_t ws_size,
                              hipStream_t stream) {
  const float* x      = (const float*)d_in[0];
  const float* w_qkv  = (const float*)d_in[1];
  const float* b_qkv  = (const float*)d_in[2];
  const float* w_g1   = (const float*)d_in[3];
  const float* b_g1   = (const float*)d_in[4];
  const float* w_g2   = (const float*)d_in[5];
  const float* b_g2   = (const float*)d_in[6];
  const float* dw_k   = (const float*)d_in[7];
  const float* dw_b   = (const float*)d_in[8];
  const float* w_proj = (const float*)d_in[9];
  const float* b_proj = (const float*)d_in[10];
  float* out = (float*)d_out;

  char* wsb = (char*)d_ws;
  size_t off = 0;
  auto take = [&](size_t bytes) -> void* {
    void* p = wsb + off;
    off = (off + bytes + 255) & ~(size_t)255;
    return p;
  };
  _Float16* xh    = (_Float16*)take((size_t)NTOK * CC * sizeof(_Float16));
  _Float16* qh    = (_Float16*)take((size_t)NTOK * CC * sizeof(_Float16));
  _Float16* kh    = (_Float16*)take((size_t)NTOK * CC * sizeof(_Float16));
  _Float16* vh    = (_Float16*)take((size_t)NTOK * CC * sizeof(_Float16));
  _Float16* aoh   = (_Float16*)take((size_t)NTOK * CC * sizeof(_Float16));
  _Float16* wqkvT = (_Float16*)take((size_t)768 * 256 * sizeof(_Float16));
  _Float16* wprjT = (_Float16*)take((size_t)256 * 256 * sizeof(_Float16));
  float*    xr    = (float*)   take((size_t)CB * CS * CC * sizeof(float));
  float*    qr    = (float*)   take((size_t)CB * CS * CC * sizeof(float));
  float*    kr    = (float*)   take((size_t)CB * CS * CC * sizeof(float));
  int*      tk    = (int*)     take((size_t)CB * CS * 4 * sizeof(int));
  float*    gw    = (float*)   take((size_t)CB * CS * 4 * sizeof(float));

  k_cvt        <<<dim3(NTOK * CC / 2048), dim3(256), 0, stream>>>(x, xh);
  k_wtrans     <<<dim3(768), dim3(256), 0, stream>>>(w_qkv, wqkvT, 256, 768);
  k_wtrans     <<<dim3(256), dim3(256), 0, stream>>>(w_proj, wprjT, 256, 256);
  k_region_mean<<<dim3(CB * CS), dim3(256), 0, stream>>>(x, xr);
  k_region_qkv <<<dim3(1024),    dim3(256), 0, stream>>>(xr, w_qkv, b_qkv, qr, kr);
  k_route      <<<dim3(CB * CS), dim3(64),  0, stream>>>(qr, kr, w_g1, b_g1, w_g2, b_g2, tk, gw);
  k_qkv_gemm   <<<dim3(NTOK / 64, 12), dim3(256), 0, stream>>>(xh, wqkvT, b_qkv, qh, kh, vh);
  k_attn       <<<dim3(CB * CS), dim3(128), 0, stream>>>(qh, kh, vh, tk, gw, aoh);
  k_dwconv     <<<dim3(CB * CS), dim3(256), 0, stream>>>(vh, dw_k, dw_b, aoh);
  k_proj       <<<dim3(NTOK / 64, 4), dim3(256), 0, stream>>>(aoh, wprjT, b_proj, out);
}
